// DeepFM_77120432766994
// MI455X (gfx1250) — matrix operand order repaired and checked by reading, not compiled, not run on hardware
//
#include <hip/hip_runtime.h>
#include <math.h>

// ---------------------------------------------------------------- types ----
typedef _Float16 __attribute__((ext_vector_type(16))) v16h;
typedef float    __attribute__((ext_vector_type(8)))  v8f;

// ------------------------------------------------------------- constants ---
#define B_ROWS   16384
#define NFIELD   10
#define EMB      32
#define DENSE    16
#define IN_DIM   336          // 10*32 + 16
#define IN_PAD   352          // padded to multiple of 32 (WMMA K-chunk)
#define N0       256
#define N1       128
#define N2       64
#define BN_INV   0.99999500003749968f   // 1/sqrt(1 + 1e-5)

// cumsum of FIELD_DIMS
__constant__ int c_field_off[NFIELD] = {
    0, 1000000, 1100000, 1101008, 1102012,
    1102114, 1103114, 1103614, 1103664, 1103776
};

// ---------------------------------------------------------------------------
// Fold BatchNorm (eval) into f16 weights:  W'[n][k] = W[n][k]*g[n]*inv,
// b'[n] = b[n]*g[n]*inv + be[n].  K zero-padded to KP.
// ---------------------------------------------------------------------------
__global__ void prep_kernel(const float* __restrict__ W, const float* __restrict__ b,
                            const float* __restrict__ g, const float* __restrict__ be,
                            _Float16* __restrict__ Wf, float* __restrict__ bf,
                            int N, int K, int KP)
{
    int tid = blockIdx.x * blockDim.x + threadIdx.x;
    int total = N * KP;
    for (int i = tid; i < total; i += gridDim.x * blockDim.x) {
        int n = i / KP, k = i % KP;
        float w = (k < K) ? W[n * K + k] * g[n] * BN_INV : 0.0f;
        Wf[i] = (_Float16)w;
    }
    if (tid < N) bf[tid] = b[tid] * g[tid] * BN_INV + be[tid];
}

// ---------------------------------------------------------------------------
// Gather + FM.  One wave per sample; lane = embedding dim.  Writes the padded
// f16 activation row and fm_out[row].  Row loads are 128B contiguous -> good
// HBM efficiency; 10 random rows/sample dominates (table > L2).
// ---------------------------------------------------------------------------
__global__ __launch_bounds__(256) void gather_fm_kernel(
    const int*   __restrict__ sparse,   // [B,10] (harness delivers int32)
    const float* __restrict__ dense,    // [B,16]
    const float* __restrict__ emb,      // [V,32]
    const float* __restrict__ bias,     // [V]
    const float* __restrict__ fm_bias,  // [1]
    _Float16*    __restrict__ xf,       // [B, IN_PAD]
    float*       __restrict__ fm)       // [B]
{
    const int wave = threadIdx.x >> 5;
    const int lane = threadIdx.x & 31;
    const int row  = blockIdx.x * 8 + wave;
    if (row >= B_ROWS) return;

    _Float16* xr = xf + (size_t)row * IN_PAD;
    float s = 0.0f, sq = 0.0f, first = 0.0f;

    #pragma unroll
    for (int f = 0; f < NFIELD; ++f) {
        int idx  = sparse[row * NFIELD + f] + c_field_off[f];
        float e  = emb[(size_t)idx * EMB + lane];   // 128B coalesced per field
        s  += e;
        sq += e * e;
        first += bias[idx];                         // uniform broadcast load
        xr[f * EMB + lane] = (_Float16)e;
    }
    if (lane < DENSE)            xr[NFIELD * EMB + lane] = (_Float16)dense[row * DENSE + lane];
    if (lane < IN_PAD - IN_DIM)  xr[IN_DIM + lane]       = (_Float16)0.0f;

    float v = s * s - sq;                           // per-dim (sum^2 - sumsq)
    #pragma unroll
    for (int off = 16; off > 0; off >>= 1) v += __shfl_xor(v, off, 32);
    if (lane == 0) fm[row] = fm_bias[0] + first + 0.5f * v;
}

// ---------------------------------------------------------------------------
// Fused 3-layer MLP + output head.  8 waves/block, 16 rows/wave, 128 rows/blk.
// A-fragment: ISA 16-bit A layout (lane<16: K{c..c+7, c+16..c+23}; lane>=16:
// K{c+8..c+15, c+24..c+31}).  B-fragment: lane = column, same K order as A.
// f32 accumulate, f16 inter-layer activations via per-wave LDS.
// ---------------------------------------------------------------------------
#define LOAD_A(dst, PTR, STRIDE, KBASE)                                         \
    { const _Float16* _p = (PTR) + r * (STRIDE) + (KBASE) + kh * 8;             \
      _Pragma("unroll") for (int _i = 0; _i < 8; ++_i) dst[_i]     = _p[_i];    \
      _Pragma("unroll") for (int _i = 0; _i < 8; ++_i) dst[8 + _i] = _p[16+_i]; }

#define LOAD_B(dst, WPTR, KP, NBASE, KBASE)                                     \
    { const _Float16* _p = (WPTR) + ((NBASE) + r) * (KP) + (KBASE) + kh * 8;    \
      _Pragma("unroll") for (int _i = 0; _i < 8; ++_i) dst[_i]     = _p[_i];    \
      _Pragma("unroll") for (int _i = 0; _i < 8; ++_i) dst[8 + _i] = _p[16+_i]; }

__global__ __launch_bounds__(256) void mlp_kernel(
    const _Float16* __restrict__ xf,                        // [B, IN_PAD]
    const float*    __restrict__ fm,                        // [B]
    const _Float16* __restrict__ W0f, const float* __restrict__ b0f,
    const _Float16* __restrict__ W1f, const float* __restrict__ b1f,
    const _Float16* __restrict__ W2f, const float* __restrict__ b2f,
    const float*    __restrict__ Wo,  const float* __restrict__ bo,
    float*          __restrict__ out)
{
    __shared__ _Float16 hbuf[8 * 16 * N0];                  // 64 KB, 8KB/wave
    const int wave = threadIdx.x >> 5;
    const int lane = threadIdx.x & 31;
    const int r    = lane & 15;                             // row / column in tile
    const int kh   = lane >> 4;                             // lane half
    const int row0 = blockIdx.x * 128 + wave * 16;
    _Float16* hw   = hbuf + wave * (16 * N0);

    // ---- Layer 0: [16,352] x [352,256] -----------------------------------
    v8f acc0[16];
    #pragma unroll
    for (int t = 0; t < 16; ++t) acc0[t] = {};
    const _Float16* xrow = xf + (size_t)row0 * IN_PAD;
    for (int kc = 0; kc < IN_PAD; kc += 32) {
        v16h a; LOAD_A(a, xrow, IN_PAD, kc);
        #pragma unroll
        for (int t = 0; t < 16; ++t) {
            v16h bm; LOAD_B(bm, W0f, IN_PAD, t * 16, kc);
            acc0[t] = __builtin_amdgcn_wmma_f32_16x16x32_f16(
                false, a, false, bm, (short)0, acc0[t], false, false);
        }
    }
    #pragma unroll
    for (int t = 0; t < 16; ++t) {
        int col = t * 16 + r;
        float bb = b0f[col];
        #pragma unroll
        for (int v = 0; v < 8; ++v) {
            float val = acc0[t][v] + bb;
            val = val > 0.0f ? val : 0.0f;                  // ReLU
            hw[(v + kh * 8) * N0 + col] = (_Float16)val;
        }
    }

    // ---- Layer 1: [16,256] x [256,128] -----------------------------------
    v8f acc1[8];
    #pragma unroll
    for (int t = 0; t < 8; ++t) acc1[t] = {};
    for (int kc = 0; kc < N0; kc += 32) {
        v16h a; LOAD_A(a, hw, N0, kc);                      // ds_load from LDS
        #pragma unroll
        for (int t = 0; t < 8; ++t) {
            v16h bm; LOAD_B(bm, W1f, N0, t * 16, kc);
            acc1[t] = __builtin_amdgcn_wmma_f32_16x16x32_f16(
                false, a, false, bm, (short)0, acc1[t], false, false);
        }
    }
    #pragma unroll
    for (int t = 0; t < 8; ++t) {                           // h0 fully consumed:
        int col = t * 16 + r;                               // reuse region for h1
        float bb = b1f[col];
        #pragma unroll
        for (int v = 0; v < 8; ++v) {
            float val = acc1[t][v] + bb;
            val = val > 0.0f ? val : 0.0f;
            hw[(v + kh * 8) * N1 + col] = (_Float16)val;
        }
    }

    // ---- Layer 2: [16,128] x [128,64] ------------------------------------
    v8f acc2[4];
    #pragma unroll
    for (int t = 0; t < 4; ++t) acc2[t] = {};
    for (int kc = 0; kc < N1; kc += 32) {
        v16h a; LOAD_A(a, hw, N1, kc);
        #pragma unroll
        for (int t = 0; t < 4; ++t) {
            v16h bm; LOAD_B(bm, W2f, N1, t * 16, kc);
            acc2[t] = __builtin_amdgcn_wmma_f32_16x16x32_f16(
                false, a, false, bm, (short)0, acc2[t], false, false);
        }
    }
    float* h2 = (float*)hw;                                 // h1 consumed: f32 [16][64]
    #pragma unroll
    for (int t = 0; t < 4; ++t) {
        int col = t * 16 + r;
        float bb = b2f[col];
        #pragma unroll
        for (int v = 0; v < 8; ++v) {
            float val = acc2[t][v] + bb;
            val = val > 0.0f ? val : 0.0f;
            h2[(v + kh * 8) * N2 + col] = val;
        }
    }

    // ---- head: dot(h2, Wo) + bo + fm, sigmoid ----------------------------
    if (lane < 16) {
        int row = row0 + lane;
        float acc = bo[0] + fm[row];
        #pragma unroll 8
        for (int c = 0; c < N2; ++c) acc += h2[lane * N2 + c] * Wo[c];
        out[row] = 1.0f / (1.0f + __expf(-acc));
    }
}

// ---------------------------------------------------------------------------
extern "C" void kernel_launch(void* const* d_in, const int* in_sizes, int n_in,
                              void* d_out, int out_size, void* d_ws, size_t ws_size,
                              hipStream_t stream)
{
    const int*   sparse = (const int*)  d_in[0];
    const float* dense  = (const float*)d_in[1];
    const float* emb    = (const float*)d_in[2];
    const float* bias   = (const float*)d_in[3];
    const float* fmb    = (const float*)d_in[4];
    const float* Wo     = (const float*)d_in[5];
    const float* bo     = (const float*)d_in[6];
    const float* W0 = (const float*)d_in[7];
    const float* b0 = (const float*)d_in[8];
    const float* g0 = (const float*)d_in[9];
    const float* be0= (const float*)d_in[10];
    const float* W1 = (const float*)d_in[11];
    const float* b1 = (const float*)d_in[12];
    const float* g1 = (const float*)d_in[13];
    const float* be1= (const float*)d_in[14];
    const float* W2 = (const float*)d_in[15];
    const float* b2 = (const float*)d_in[16];
    const float* g2 = (const float*)d_in[17];
    const float* be2= (const float*)d_in[18];
    float* out = (float*)d_out;

    // workspace carve-out (256B aligned)
    char* ws = (char*)d_ws;
    size_t off = 0;
    auto alloc = [&](size_t bytes) -> void* {
        void* p = ws + off;
        off += (bytes + 255) & ~(size_t)255;
        return p;
    };
    _Float16* xf  = (_Float16*)alloc((size_t)B_ROWS * IN_PAD * 2);  // 11.5 MB
    float*    fm  = (float*)   alloc((size_t)B_ROWS * 4);
    _Float16* W0f = (_Float16*)alloc((size_t)N0 * IN_PAD * 2);
    float*    b0f = (float*)   alloc((size_t)N0 * 4);
    _Float16* W1f = (_Float16*)alloc((size_t)N1 * N0 * 2);
    float*    b1f = (float*)   alloc((size_t)N1 * 4);
    _Float16* W2f = (_Float16*)alloc((size_t)N2 * N1 * 2);
    float*    b2f = (float*)   alloc((size_t)N2 * 4);

    // 1) fold BN into f16 weights
    prep_kernel<<<(N0 * IN_PAD + 255) / 256, 256, 0, stream>>>(W0, b0, g0, be0, W0f, b0f, N0, IN_DIM, IN_PAD);
    prep_kernel<<<(N1 * N0     + 255) / 256, 256, 0, stream>>>(W1, b1, g1, be1, W1f, b1f, N1, N0, N0);
    prep_kernel<<<(N2 * N1     + 255) / 256, 256, 0, stream>>>(W2, b2, g2, be2, W2f, b2f, N2, N1, N1);

    // 2) embedding gather + FM (1 wave per sample)
    gather_fm_kernel<<<B_ROWS / 8, 256, 0, stream>>>(sparse, dense, emb, bias, fmb, xf, fm);

    // 3) fused WMMA MLP + head (128 rows per block)
    mlp_kernel<<<B_ROWS / 128, 256, 0, stream>>>(xf, fm, W0f, b0f, W1f, b1f, W2f, b2f, Wo, bo, out);
}
